// Attention_5480378270188
// MI455X (gfx1250) — compile-verified, matrix operand order checked by reading, not run on hardware
//
#include <hip/hip_runtime.h>
#include <hip/hip_bf16.h>

// Problem constants (match reference)
#define B_     8
#define N_     2048
#define M_     2048
#define DMODEL 1024
#define DK_    1024
#define DV_    1024
#define NEGV   (-1000000000.0f)

// LDS staging geometry for the shared (B) operand of the attention GEMMs:
// 64 rows x 256 K-elements per slice, row stride padded to 264 bf16 elements
// (528 B) so 16 lanes reading consecutive rows land on distinct bank groups.
#define KSLICE   256
#define KSTRIDE  264
#define BUFELEMS (64 * KSTRIDE)

typedef __attribute__((ext_vector_type(16))) __bf16 v16bf;
typedef __attribute__((ext_vector_type(8)))  __bf16 v8bf;
typedef __attribute__((ext_vector_type(8)))  float  v8f;
typedef int v4i __attribute__((vector_size(16)));   // matches builtin param type

// ---------------------------------------------------------------------------
// CDNA5 async global->LDS copy (ASYNCcnt path), with a portable fallback.
// Builtin signature (from hipcc diagnostic): (AS1 v4i*, AS3 v4i*, Iint, Iint).
// ---------------------------------------------------------------------------
#if defined(__HIP_DEVICE_COMPILE__) && __has_builtin(__builtin_amdgcn_global_load_async_to_lds_b128)
#define HAVE_ASYNC_LDS 1
#endif

static __device__ inline void cp16_g2lds(__bf16* lds, const __bf16* g) {
#if HAVE_ASYNC_LDS
    __builtin_amdgcn_global_load_async_to_lds_b128(
        (__attribute__((address_space(1))) v4i*)(g),
        (__attribute__((address_space(3))) v4i*)(lds),
        0, 0);
#else
    *reinterpret_cast<v8bf*>(lds) = *reinterpret_cast<const v8bf*>(g);
#endif
}

// Wait until at most the last-issued stage (8 async copies per wave) is
// still in flight.  Async loads complete in order, so ASYNCcnt <= 8 proves
// the previous stage's copies have landed in LDS.
static __device__ inline void cp_wait_prev() {
#if HAVE_ASYNC_LDS
    asm volatile("s_wait_asynccnt 0x8" ::: "memory");
#endif
}
static __device__ inline void cp_wait_all() {
#if HAVE_ASYNC_LDS
    asm volatile("s_wait_asynccnt 0x0" ::: "memory");
#endif
}

// ---------------------------------------------------------------------------
// Fragment builders.
// A-matrix (16x32 bf16) per-lane layout (ISA 05_wmma §7.12.2):
//   lane L (both halves) holds row M = L&15.
//   elems [0..7]  = K (k0 .. k0+7)
//   elems [8..15] = K (k0+16 .. k0+23)
//   where k0 = ktile + (lane>=16 ? 8 : 0).
// B operand is loaded from B^T rows (K contiguous) with the mirrored layout,
// so both GEMM operands of an A·B^T product load identically.
// ---------------------------------------------------------------------------

static __device__ inline v16bf frag_from_f32(const float* __restrict__ p, int k0) {
    float4 a0 = *reinterpret_cast<const float4*>(p + k0);
    float4 a1 = *reinterpret_cast<const float4*>(p + k0 + 4);
    float4 b0 = *reinterpret_cast<const float4*>(p + k0 + 16);
    float4 b1 = *reinterpret_cast<const float4*>(p + k0 + 20);
    v16bf f;
    f[0]  = (__bf16)a0.x; f[1]  = (__bf16)a0.y; f[2]  = (__bf16)a0.z; f[3]  = (__bf16)a0.w;
    f[4]  = (__bf16)a1.x; f[5]  = (__bf16)a1.y; f[6]  = (__bf16)a1.z; f[7]  = (__bf16)a1.w;
    f[8]  = (__bf16)b0.x; f[9]  = (__bf16)b0.y; f[10] = (__bf16)b0.z; f[11] = (__bf16)b0.w;
    f[12] = (__bf16)b1.x; f[13] = (__bf16)b1.y; f[14] = (__bf16)b1.z; f[15] = (__bf16)b1.w;
    return f;
}

// Works for both global bf16 (global_load_b128) and LDS bf16 (ds_load_b128).
static __device__ inline v16bf frag_from_bf16(const __bf16* p, int k0) {
    v8bf lo = *reinterpret_cast<const v8bf*>(p + k0);
    v8bf hi = *reinterpret_cast<const v8bf*>(p + k0 + 16);
    v16bf f;
#pragma unroll
    for (int i = 0; i < 8; ++i) { f[i] = lo[i]; f[8 + i] = hi[i]; }
    return f;
}

// ---------------------------------------------------------------------------
// Stage 1: Y = X @ W^T + bias   (fp32 in, bf16 out, fp32 accumulate)
// transposeV==0: Y[row*1024 + col]  (Q, K)
// transposeV==1: Y[(b*DV + col)*M + m]  (V stored transposed per batch)
// Block: 256 threads (8 waves). Tile: 128 rows x 64 cols, k-step 32.
// ---------------------------------------------------------------------------
__global__ __launch_bounds__(256) void proj_kernel(
    const float* __restrict__ X, const float* __restrict__ W,
    const float* __restrict__ bias, __bf16* __restrict__ Y, int transposeV)
{
    const int lane  = threadIdx.x & 31;
    const int wave  = threadIdx.x >> 5;
    const int r0    = blockIdx.x * 128 + wave * 16;
    const int c0    = blockIdx.y * 64;
    const int lrow  = r0 + (lane & 15);
    const int half8 = (lane >> 4) * 8;

    v8f acc[4];
#pragma unroll
    for (int t = 0; t < 4; ++t) acc[t] = (v8f){0.f,0.f,0.f,0.f,0.f,0.f,0.f,0.f};

    const float* xrow = X + (size_t)lrow * DMODEL;
    for (int kk = 0; kk < DMODEL; kk += 32) {
        const int k0 = kk + half8;
        v16bf af = frag_from_f32(xrow, k0);
#pragma unroll
        for (int t = 0; t < 4; ++t) {
            const int col = c0 + t * 16 + (lane & 15);
            v16bf bf = frag_from_f32(W + (size_t)col * DMODEL, k0);
            acc[t] = __builtin_amdgcn_wmma_f32_16x16x32_bf16(
                false, af, false, bf, (short)0, acc[t], false, false);
        }
    }

#pragma unroll
    for (int t = 0; t < 4; ++t) {
        const int col = c0 + t * 16 + (lane & 15);
        const float bv = bias[col];
#pragma unroll
        for (int r = 0; r < 8; ++r) {
            const int row = r0 + r + half8;
            const float v = acc[t][r] + bv;
            if (!transposeV) {
                Y[(size_t)row * 1024 + col] = (__bf16)v;
            } else {
                const int bidx = row >> 11;           // row / M_
                const int m    = row & (M_ - 1);      // row % M_
                Y[((size_t)bidx * DV_ + col) * M_ + m] = (__bf16)v;
            }
        }
    }
}

// ---------------------------------------------------------------------------
// Stage 2: S = Q @ K^T (bf16 WMMA, fp32 accum), mask applied, raw S stored.
// Per-row online softmax stats (m, l) -> stats[(b*N + row)*2 + {0,1}].
// Grid: (N/128, B). Each wave owns 16 Q rows x full M (64-col chunks).
// The shared K operand is staged to LDS with DOUBLE-BUFFERED async copies:
// a 128-stage software pipeline (32 col-chunks x 4 K-slices) that issues the
// next 32 KB slice before waiting on the current one (s_wait_asynccnt 0x8).
// ---------------------------------------------------------------------------
__global__ __launch_bounds__(256) void scores_kernel(
    const __bf16* __restrict__ Q, const __bf16* __restrict__ K,
    const int* __restrict__ mask, float* __restrict__ S,
    float* __restrict__ stats)
{
    __shared__ __bf16 kbuf[2][BUFELEMS];   // 2 x 33.8 KB

    const int tid   = threadIdx.x;
    const int lane  = tid & 31;
    const int wave  = tid >> 5;
    const int b     = blockIdx.y;
    const int r0    = blockIdx.x * 128 + wave * 16;
    const int lrow  = r0 + (lane & 15);
    const int half8 = (lane >> 4) * 8;

    const __bf16* qrow = Q + ((size_t)b * N_ + lrow) * DK_;

    // Issue the 8 async b128 copies of pipeline stage `st` (2048 segments,
    // 8 per thread) into kbuf[st&1].  Stage -> (mc, ks) = (st/4*64, st%4*256).
    auto issue_stage = [&](int st) {
        const int mc2 = (st >> 2) * 64;
        const int ks2 = (st & 3) * KSLICE;
        __bf16* dst = kbuf[st & 1];
#pragma unroll
        for (int i = 0; i < 8; ++i) {
            const int s    = tid + 256 * i;
            const int row  = s >> 5;          // 32 segs per row
            const int kseg = s & 31;
            cp16_g2lds(dst + row * KSTRIDE + kseg * 8,
                       K + ((size_t)b * M_ + mc2 + row) * DK_ + ks2 + kseg * 8);
        }
    };

    float mrun[8], lrun[8];
#pragma unroll
    for (int r = 0; r < 8; ++r) { mrun[r] = -3.0e38f; lrun[r] = 0.0f; }

    v8f acc[4];
    const int NSTG = (M_ / 64) * (DK_ / KSLICE);   // 128 stages

    issue_stage(0);
    for (int st = 0; st < NSTG; ++st) {
        const int mc  = (st >> 2) * 64;
        const int buf = st & 1;

        if (st + 1 < NSTG) { issue_stage(st + 1); cp_wait_prev(); }
        else               { cp_wait_all(); }
        __syncthreads();    // all waves' copies for stage `st` have landed

        if ((st & 3) == 0) {
#pragma unroll
            for (int t = 0; t < 4; ++t)
                acc[t] = (v8f){0.f,0.f,0.f,0.f,0.f,0.f,0.f,0.f};
        }

        const int ks = (st & 3) * KSLICE;
        for (int kk = 0; kk < KSLICE; kk += 32) {
            const int k0l = kk + half8;
            v16bf af = frag_from_bf16(qrow, ks + k0l);
#pragma unroll
            for (int t = 0; t < 4; ++t) {
                const int cl = t * 16 + (lane & 15);
                v16bf bf = frag_from_bf16(&kbuf[buf][cl * KSTRIDE], k0l);
                acc[t] = __builtin_amdgcn_wmma_f32_16x16x32_bf16(
                    false, af, false, bf, (short)0, acc[t], false, false);
            }
        }
        __syncthreads();    // readers done before this buffer is re-filled

        if ((st & 3) == 3) {
            // ---- chunk epilogue: mask, store raw S, online softmax stats ----
            float mx[8];
#pragma unroll
            for (int r = 0; r < 8; ++r) mx[r] = -3.0e38f;
#pragma unroll
            for (int t = 0; t < 4; ++t) {
                const int col = mc + t * 16 + (lane & 15);
#pragma unroll
                for (int r = 0; r < 8; ++r) {
                    const int row = r0 + r + half8;
                    float s = acc[t][r];
                    if (mask[(size_t)row * M_ + col] == 0) s = NEGV;
                    acc[t][r] = s;
                    S[((size_t)b * N_ + row) * M_ + col] = s;
                    mx[r] = fmaxf(mx[r], s);
                }
            }
#pragma unroll
            for (int r = 0; r < 8; ++r) {
                float v = mx[r];
#pragma unroll
                for (int off = 1; off < 16; off <<= 1)
                    v = fmaxf(v, __shfl_xor(v, off, 32));
                mx[r] = v;
            }
#pragma unroll
            for (int r = 0; r < 8; ++r) {
                const float mnew = fmaxf(mrun[r], mx[r]);
                float se = 0.0f;
#pragma unroll
                for (int t = 0; t < 4; ++t) se += __expf(acc[t][r] - mnew);
#pragma unroll
                for (int off = 1; off < 16; off <<= 1)
                    se += __shfl_xor(se, off, 32);
                lrun[r] = lrun[r] * __expf(mrun[r] - mnew) + se;
                mrun[r] = mnew;
            }
        }
    }

    if ((lane & 15) == 0) {
#pragma unroll
        for (int r = 0; r < 8; ++r) {
            const int row = r0 + r + half8;
            const size_t o = ((size_t)b * N_ + row) * 2;
            stats[o]     = mrun[r];
            stats[o + 1] = lrun[r];
        }
    }
}

// ---------------------------------------------------------------------------
// Stage 3: O = softmax(S) @ V.  P built on the fly: p = exp(s - m_row)/l_row,
// cast bf16 into A fragments; B fragments from V^T staged through LDS with
// double-buffered async copies (8-stage pipeline over m).
// Grid: (N/128, DV/64, B). fp32 accumulate, fp32 store to d_out.
// ---------------------------------------------------------------------------
__global__ __launch_bounds__(256) void outgemm_kernel(
    const float* __restrict__ S, const float* __restrict__ stats,
    const __bf16* __restrict__ Vt, float* __restrict__ O)
{
    __shared__ __bf16 vbuf[2][BUFELEMS];   // 2 x 33.8 KB

    const int tid   = threadIdx.x;
    const int lane  = tid & 31;
    const int wave  = tid >> 5;
    const int b     = blockIdx.z;
    const int r0    = blockIdx.x * 128 + wave * 16;
    const int c0    = blockIdx.y * 64;
    const int lrow  = r0 + (lane & 15);
    const int half8 = (lane >> 4) * 8;

    const size_t so = ((size_t)b * N_ + lrow) * 2;
    const float mrow = stats[so];
    const float invl = 1.0f / stats[so + 1];
    const float* srow = S + ((size_t)b * N_ + lrow) * M_;

    auto issue_stage = [&](int st) {
        const int ms2 = st * KSLICE;
        __bf16* dst = vbuf[st & 1];
#pragma unroll
        for (int i = 0; i < 8; ++i) {
            const int s    = tid + 256 * i;
            const int row  = s >> 5;              // dv-col-local 0..63
            const int kseg = s & 31;
            cp16_g2lds(dst + row * KSTRIDE + kseg * 8,
                       Vt + ((size_t)b * DV_ + c0 + row) * M_ + ms2 + kseg * 8);
        }
    };

    v8f acc[4];
#pragma unroll
    for (int t = 0; t < 4; ++t) acc[t] = (v8f){0.f,0.f,0.f,0.f,0.f,0.f,0.f,0.f};

    const int NSTG = M_ / KSLICE;   // 8 stages

    issue_stage(0);
    for (int st = 0; st < NSTG; ++st) {
        const int ms  = st * KSLICE;
        const int buf = st & 1;

        if (st + 1 < NSTG) { issue_stage(st + 1); cp_wait_prev(); }
        else               { cp_wait_all(); }
        __syncthreads();

        for (int mm = 0; mm < KSLICE; mm += 32) {
            const int k0  = ms + mm + half8;      // global m offset for A
            const int k0l = mm + half8;           // LDS-local offset for B
            float4 a0 = *reinterpret_cast<const float4*>(srow + k0);
            float4 a1 = *reinterpret_cast<const float4*>(srow + k0 + 4);
            float4 b0 = *reinterpret_cast<const float4*>(srow + k0 + 16);
            float4 b1 = *reinterpret_cast<const float4*>(srow + k0 + 20);
            v16bf af;
            af[0]  = (__bf16)(__expf(a0.x - mrow) * invl);
            af[1]  = (__bf16)(__expf(a0.y - mrow) * invl);
            af[2]  = (__bf16)(__expf(a0.z - mrow) * invl);
            af[3]  = (__bf16)(__expf(a0.w - mrow) * invl);
            af[4]  = (__bf16)(__expf(a1.x - mrow) * invl);
            af[5]  = (__bf16)(__expf(a1.y - mrow) * invl);
            af[6]  = (__bf16)(__expf(a1.z - mrow) * invl);
            af[7]  = (__bf16)(__expf(a1.w - mrow) * invl);
            af[8]  = (__bf16)(__expf(b0.x - mrow) * invl);
            af[9]  = (__bf16)(__expf(b0.y - mrow) * invl);
            af[10] = (__bf16)(__expf(b0.z - mrow) * invl);
            af[11] = (__bf16)(__expf(b0.w - mrow) * invl);
            af[12] = (__bf16)(__expf(b1.x - mrow) * invl);
            af[13] = (__bf16)(__expf(b1.y - mrow) * invl);
            af[14] = (__bf16)(__expf(b1.z - mrow) * invl);
            af[15] = (__bf16)(__expf(b1.w - mrow) * invl);
#pragma unroll
            for (int t = 0; t < 4; ++t) {
                const int cl = t * 16 + (lane & 15);
                v16bf bf = frag_from_bf16(&vbuf[buf][cl * KSTRIDE], k0l);
                acc[t] = __builtin_amdgcn_wmma_f32_16x16x32_bf16(
                    false, af, false, bf, (short)0, acc[t], false, false);
            }
        }
        __syncthreads();   // readers done before this buffer is re-filled
    }

#pragma unroll
    for (int t = 0; t < 4; ++t) {
        const int col = c0 + t * 16 + (lane & 15);
#pragma unroll
        for (int r = 0; r < 8; ++r) {
            const int row = r0 + r + half8;
            O[((size_t)b * N_ + row) * DV_ + col] = acc[t][r];
        }
    }
}

// ---------------------------------------------------------------------------
extern "C" void kernel_launch(void* const* d_in, const int* in_sizes, int n_in,
                              void* d_out, int out_size, void* d_ws, size_t ws_size,
                              hipStream_t stream) {
    const float* querys = (const float*)d_in[0];
    const float* keys   = (const float*)d_in[1];
    const float* values = (const float*)d_in[2];
    const int*   mask   = (const int*)d_in[3];
    const float* Wq     = (const float*)d_in[4];
    const float* bq     = (const float*)d_in[5];
    const float* Wk     = (const float*)d_in[6];
    const float* bk     = (const float*)d_in[7];
    const float* Wv     = (const float*)d_in[8];
    const float* bv     = (const float*)d_in[9];
    float* out = (float*)d_out;

    const size_t nTok = (size_t)B_ * N_;          // 16384 tokens
    char* ws = (char*)d_ws;
    __bf16* qb = (__bf16*)ws;  ws += nTok * DK_ * sizeof(__bf16);   // 32 MB
    __bf16* kb = (__bf16*)ws;  ws += nTok * DK_ * sizeof(__bf16);   // 32 MB
    __bf16* vT = (__bf16*)ws;  ws += nTok * DV_ * sizeof(__bf16);   // 32 MB
    float*  S  = (float*)ws;   ws += (size_t)B_ * N_ * M_ * sizeof(float); // 128 MB
    float*  st = (float*)ws;   // [B*N, 2] row stats (m, l)

    dim3 blk(256);
    dim3 gproj(nTok / 128, 1024 / 64);            // 128 x 16
    proj_kernel<<<gproj, blk, 0, stream>>>(querys, Wq, bq, qb, 0);
    proj_kernel<<<gproj, blk, 0, stream>>>(keys,   Wk, bk, kb, 0);
    proj_kernel<<<gproj, blk, 0, stream>>>(values, Wv, bv, vT, 1);

    dim3 gsc(N_ / 128, B_);                       // 16 x 8
    scores_kernel<<<gsc, blk, 0, stream>>>(qb, kb, mask, S, st);

    dim3 gout(N_ / 128, DV_ / 64, B_);            // 16 x 16 x 8
    outgemm_kernel<<<gout, blk, 0, stream>>>(S, st, vT, out);
}
